// HybridLoss_88665304858861
// MI455X (gfx1250) — matrix operand order repaired and checked by reading, not compiled, not run on hardware
//
#include <hip/hip_runtime.h>

typedef _Float16 half_t;
typedef __attribute__((ext_vector_type(16))) _Float16 v16h;
typedef __attribute__((ext_vector_type(8)))  _Float16 v8h;
typedef __attribute__((ext_vector_type(8)))  float    v8f;

#define IMG_H 512
#define IMG_W 512
#define NIMG  96                      // 32 batch * 3 channels
#define TILE_W 64
#define TILE_H 32
#define TILES_X (IMG_W / TILE_W)      // 8
#define TILES_Y (IMG_H / TILE_H)      // 16
#define TILES_PER_IMG (TILES_X * TILES_Y)   // 128
#define NWG (NIMG * TILES_PER_IMG)          // 12288
#define IN_H 48                       // 32 + 10 halo, padded to 48 (A rows reach 47)
#define IN_W 80                       // 64 + 10 halo, padded to 80 (A cols reach 79)
#define NFIELDS 5
#define NPIX 25165824.0f              // 32*3*512*512

// Normalized 1-D Gaussian, win=11, sigma=1.5 (outer-product kernel is separable)
__device__ __constant__ float G11[11] = {
    0.00102838f, 0.00759876f, 0.03600072f, 0.10936076f, 0.21300577f,
    0.26601150f,
    0.21300577f, 0.10936076f, 0.03600072f, 0.00759876f, 0.00102838f};

__device__ __forceinline__ float gband(int d) {
  return (d >= 0 && d <= 10) ? G11[d] : 0.0f;
}

// ---- WMMA fragment helpers (CDNA5 wave32 layouts, CDNA5 ISA) ----

// 16-bit A matrix 16x32: M = lane&15; lanes 0-15 kb=0, lanes 16-31 kb=8;
// elements = two contiguous 8-half slabs at columns [kb, kb+8) and [16+kb, 16+kb+8).
__device__ __forceinline__ v16h load_a_frag(const half_t* base, int stride,
                                            int rowBase, int colBase, int lane) {
  const int M  = lane & 15;
  const int kb = (lane < 16) ? 0 : 8;
  const half_t* rp = base + (rowBase + M) * stride + colBase;
  const v8h lo = *reinterpret_cast<const v8h*>(rp + kb);        // 16B aligned
  const v8h hi = *reinterpret_cast<const v8h*>(rp + 16 + kb);   // 16B aligned
  v16h a;
#pragma unroll
  for (int e = 0; e < 8; ++e) { a[e] = lo[e]; a[8 + e] = hi[e]; }
  return a;
}

// B-fragment-ordered buffer: bLane = (K>=16 ? 16 : 0) + N owns elements e = K&15,
// stored contiguously at buf[bLane*16 + e].  Consumer: one aligned 32B load.
__device__ __forceinline__ v16h load_b_frag_contig(const half_t* buf, int lane) {
  return *reinterpret_cast<const v16h*>(buf + lane * 16);
}

// f32 C/D 16x16 tile (rows kOff..kOff+15) -> B-fragment-ordered f16 buffer.
// Lane holds col N, rows K = kOff + rb + r; targets are contiguous: one b128 store.
__device__ __forceinline__ void store_c_bfrag(half_t* buf, const v8f c,
                                              int kOff, int lane) {
  const int N  = lane & 15;
  const int rb = (lane < 16) ? 0 : 8;
  v8h h;
#pragma unroll
  for (int r = 0; r < 8; ++r) h[r] = (half_t)c[r];
  *reinterpret_cast<v8h*>(buf + (kOff + N) * 16 + rb) = h;      // 16B aligned
}

__global__ __launch_bounds__(256)
void hybrid_loss_tile(const float* __restrict__ pred,
                      const float* __restrict__ targ,
                      float* __restrict__ partials) {
  __shared__ half_t s_fld[NFIELDS][IN_H * IN_W];      // pc, t, pc^2, t^2, pc*t
  __shared__ __align__(32) half_t s_th[8][32 * 16];   // per-wave staging, B-frag order
  __shared__ __align__(32) half_t s_ghf[32 * 16];     // Gh fragments, pre-swizzled
  __shared__ __align__(32) half_t s_gvf[32 * 16];     // Gv fragments, pre-swizzled
  __shared__ float  s_r0[256];
  __shared__ float  s_r1[256];

  const int t    = threadIdx.x;
  const int lane = t & 31;
  const int wave = t >> 5;
  const int wg   = blockIdx.x;

  const int img  = wg / TILES_PER_IMG;
  const int tidx = wg - img * TILES_PER_IMG;
  const int tx   = tidx & (TILES_X - 1);
  const int ty   = tidx >> 3;            // TILES_X == 8
  const int x0   = tx * TILE_W;
  const int y0   = ty * TILE_H;
  const size_t ib = (size_t)img * IMG_H * IMG_W;

  // Pre-swizzle the banded Gaussian operators straight into fragment order.
  for (int i = t; i < 512; i += 256) {
    // Gh (B operand, 32x16): Gh[K][N] = g[K-N]; lane half h=bl>>4, e<8: K=8h+e, e>=8: K=16+8h+(e-8).
    {
      const int bl = i >> 4, e = i & 15;
      const int N  = bl & 15;
      const int K  = ((bl < 16) ? 0 : 8) + ((e < 8) ? e : 8 + e);
      s_ghf[i] = (half_t)gband(K - N);
    }
    // Gv (A operand, 16x32): Gv[M][K] = g[K-M]; K = (al>=16)*16 + e (matches staged Th).
    {
      const int al = i >> 4, e = i & 15;
      const int M  = al & 15;
      const int kb = (al < 16) ? 0 : 8;
      const int v  = e >> 1, p = e & 1;
      const int K  = ((al < 16) ? 0 : 16) + e;
      s_gvf[i] = (half_t)gband(K - M);
    }
  }

  // Cooperative halo load: zero-pad outside image AND in the pad lanes so no
  // undefined LDS (possible NaN bit patterns) ever feeds a WMMA.
  float mse_acc = 0.0f;
  for (int i = t; i < IN_H * IN_W; i += 256) {
    const int r  = i / IN_W, c = i - r * IN_W;
    const int gy = y0 - 5 + r, gx = x0 - 5 + c;
    float p = 0.0f, q = 0.0f;
    if (r < TILE_H + 10 && c < TILE_W + 10 &&
        gy >= 0 && gy < IMG_H && gx >= 0 && gx < IMG_W) {
      p = pred[ib + (size_t)gy * IMG_W + gx];
      q = targ[ib + (size_t)gy * IMG_W + gx];
    }
    // MSE over uniquely-owned interior pixels, with UNclipped pred (reference).
    if (r >= 5 && r < 5 + TILE_H && c >= 5 && c < 5 + TILE_W) {
      const float d = p - q;
      mse_acc += d * d;
    }
    const float pc = fminf(fmaxf(p, 0.0f), 1.0f);  // SSIM clips pred to [0,1]
    s_fld[0][i] = (half_t)pc;
    s_fld[1][i] = (half_t)q;
    s_fld[2][i] = (half_t)(pc * pc);
    s_fld[3][i] = (half_t)(q * q);
    s_fld[4][i] = (half_t)(pc * q);
  }
  __syncthreads();

  // Each wave owns one 16x16 output sub-tile of the 64x32 tile.
  const int r0 = (wave >> 2) * 16;   // input-row offset (0 or 16)
  const int c0 = (wave & 3) * 16;    // output-col offset (0/16/32/48)
  half_t* myth = s_th[wave];

  const v16h ghB = load_b_frag_contig(s_ghf, lane);
  const v16h gvA = load_b_frag_contig(s_gvf, lane);   // same contiguous packing

  v8f res[NFIELDS];
#pragma unroll
  for (int f = 0; f < NFIELDS; ++f) {
    const half_t* fld = s_fld[f];
    // Horizontal pass: Th(32x16) = In(32x32) x Gh(32x16), two 16-row WMMAs.
    const v16h a0 = load_a_frag(fld, IN_W, r0,      c0, lane);
    const v16h a1 = load_a_frag(fld, IN_W, r0 + 16, c0, lane);
    v8f z = {};
    v8f th0 = __builtin_amdgcn_wmma_f32_16x16x32_f16(false, a0, false, ghB,
                                                     (short)0, z, false, false);
    v8f th1 = __builtin_amdgcn_wmma_f32_16x16x32_f16(false, a1, false, ghB,
                                                     (short)0, z, false, false);
    // C -> B re-layout via per-wave LDS staging, one b128 store per lane per tile.
    store_c_bfrag(myth, th0, 0,  lane);
    store_c_bfrag(myth, th1, 16, lane);
    asm volatile("s_wait_dscnt 0x0" ::: "memory");
    const v16h tb = load_b_frag_contig(myth, lane);
    // Vertical pass: Out(16x16) = Gv(16x32) x Th(32x16).
    res[f] = __builtin_amdgcn_wmma_f32_16x16x32_f16(false, gvA, false, tb,
                                                    (short)0, z, false, false);
  }

  // SSIM map on the 8 output pixels this lane owns (C-layout rows).
  const float C1c = 0.0001f;   // (0.01*1.0)^2
  const float C2c = 0.0009f;   // (0.03*1.0)^2
  float ssim_acc = 0.0f;
#pragma unroll
  for (int r = 0; r < 8; ++r) {
    const float mx  = res[0][r], my = res[1][r];
    const float mx2 = mx * mx, my2 = my * my, mxy = mx * my;
    const float sx2 = res[2][r] - mx2;
    const float sy2 = res[3][r] - my2;
    const float sxy = res[4][r] - mxy;
    const float num = (2.0f * mxy + C1c) * (2.0f * sxy + C2c);
    const float den = (mx2 + my2 + C1c) * (sx2 + sy2 + C2c);
    ssim_acc += num / den;
  }

  // Deterministic in-block reduction, one partial pair per workgroup.
  s_r0[t] = mse_acc;
  s_r1[t] = ssim_acc;
  __syncthreads();
  for (int off = 128; off > 0; off >>= 1) {
    if (t < off) {
      s_r0[t] += s_r0[t + off];
      s_r1[t] += s_r1[t + off];
    }
    __syncthreads();
  }
  if (t == 0) {
    partials[2 * wg]     = s_r0[0];
    partials[2 * wg + 1] = s_r1[0];
  }
}

__global__ __launch_bounds__(256)
void hybrid_loss_finalize(const float* __restrict__ partials,
                          float* __restrict__ out, int n) {
  __shared__ float sm[256];
  __shared__ float ss[256];
  float m = 0.0f, s = 0.0f;
  for (int i = threadIdx.x; i < n; i += 256) {
    m += partials[2 * i];
    s += partials[2 * i + 1];
  }
  sm[threadIdx.x] = m;
  ss[threadIdx.x] = s;
  __syncthreads();
  for (int off = 128; off > 0; off >>= 1) {
    if (threadIdx.x < off) {
      sm[threadIdx.x] += sm[threadIdx.x + off];
      ss[threadIdx.x] += ss[threadIdx.x + off];
    }
    __syncthreads();
  }
  if (threadIdx.x == 0) {
    const float mse  = sm[0] / NPIX;
    const float ssim = ss[0] / NPIX;
    out[0] = 0.8f * mse + 0.2f * (1.0f - ssim);  // ALPHA*mse + (1-ALPHA)*(1-ssim)
  }
}

extern "C" void kernel_launch(void* const* d_in, const int* in_sizes, int n_in,
                              void* d_out, int out_size, void* d_ws, size_t ws_size,
                              hipStream_t stream) {
  const float* pred = (const float*)d_in[0];
  const float* targ = (const float*)d_in[1];
  float* partials = (float*)d_ws;   // NWG * 2 floats = 96 KB of scratch

  hybrid_loss_tile<<<dim3(NWG), dim3(256), 0, stream>>>(pred, targ, partials);
  hybrid_loss_finalize<<<dim3(1), dim3(256), 0, stream>>>(partials, (float*)d_out, NWG);
}
